// LinearTransformerNd_16501264352058
// MI455X (gfx1250) — compile-verified, matrix operand order checked by reading, not run on hardware
//
#include <hip/hip_runtime.h>
#include <hip/hip_bf16.h>

typedef __attribute__((ext_vector_type(16))) _Float16 v16h;
typedef __attribute__((ext_vector_type(8)))  _Float16 v8h;
typedef __attribute__((ext_vector_type(4)))  _Float16 v4h;
typedef __attribute__((ext_vector_type(8)))  float    v8f;
typedef __attribute__((ext_vector_type(4)))  float    v4f;

#define NUM_HEADS 8
#define CH        32
#define C_DIM     256
#define N_DIM     4096
#define B_DIM     2
#define EPS       1e-5f
#define ATT_SCALE 0.42044820762685725f   // 32^(-0.25)

// ---------------- WMMA fragment helpers (wave32, 16x16x32 f16) ----------------
// Assumed 16-bit fragment layout (CDNA5 ISA A-layout, B symmetric):
//   lanes 0-15  : row/col = lane,    K in {0..7} U {16..23}
//   lanes 16-31 : row/col = lane-16, K in {8..15} U {24..31}
// => per lane: two contiguous 8-half (16 B) runs at K = 8g and K = 16+8g.
__device__ __forceinline__ v16h load_frag_f16(const _Float16* __restrict__ p, int ld) {
    int lane = threadIdx.x & 31;
    int g = lane >> 4, r = lane & 15;
    const _Float16* row = p + r * ld + g * 8;
    v8h lo = *(const v8h*)(row);        // K = 8g .. 8g+7
    v8h hi = *(const v8h*)(row + 16);   // K = 16+8g .. 23+8g
    v16h out;
#pragma unroll
    for (int i = 0; i < 8; ++i) { out[i] = lo[i]; out[i + 8] = hi[i]; }
    return out;
}

#define WMMA_F16(A, B, Cacc) \
    __builtin_amdgcn_wmma_f32_16x16x32_f16(false, (A), false, (B), (short)0, (Cacc), false, false)

// ---------------- Kernel 0: one-shot f32 -> f16 weight conversion ----------------
__global__ void wcvt_kernel(const float* __restrict__ wq, const float* __restrict__ wk,
                            const float* __restrict__ wv, const float* __restrict__ wo,
                            _Float16* __restrict__ hq, _Float16* __restrict__ hk,
                            _Float16* __restrict__ hv, _Float16* __restrict__ ho) {
    int i = (blockIdx.x * 256 + threadIdx.x) * 4;   // 16384 threads * 4 = 65536 elems
    const float* src[4] = { wq, wk, wv, wo };
    _Float16*    dst[4] = { hq, hk, hv, ho };
#pragma unroll
    for (int m = 0; m < 4; ++m) {
        v4f a = *(const v4f*)(src[m] + i);
        v4h o;
#pragma unroll
        for (int j = 0; j < 4; ++j) o[j] = (_Float16)a[j];
        *(v4h*)(dst[m] + i) = o;
    }
}

// ---------------- Kernel 1: GroupNorm -> f16 xh[b][n][c] (c contiguous) ----------------
__global__ void gn_kernel(const float* __restrict__ x, const float* __restrict__ gamma,
                          const float* __restrict__ beta, _Float16* __restrict__ xh) {
    int bg = blockIdx.x;          // 0..63 : b*32 + group
    int b = bg >> 5, g = bg & 31;
    int c0 = g * 8;
    const float* xp = x + ((size_t)b * C_DIM + c0) * N_DIM;

    __shared__ float ssum[256];
    __shared__ float ssq[256];
    float s = 0.f, sq = 0.f;
#pragma unroll
    for (int cc = 0; cc < 8; ++cc) {
        for (int n = threadIdx.x; n < N_DIM; n += 256) {   // coalesced
            float v = xp[cc * N_DIM + n];
            s += v; sq += v * v;
        }
    }
    ssum[threadIdx.x] = s; ssq[threadIdx.x] = sq;
    __syncthreads();
    for (int off = 128; off > 0; off >>= 1) {
        if ((int)threadIdx.x < off) {
            ssum[threadIdx.x] += ssum[threadIdx.x + off];
            ssq[threadIdx.x]  += ssq[threadIdx.x + off];
        }
        __syncthreads();
    }
    const float inv_n = 1.0f / (8.0f * N_DIM);
    float mu  = ssum[0] * inv_n;
    float var = ssq[0] * inv_n - mu * mu;
    float rs  = rsqrtf(var + EPS);

    float gsc[8], gbi[8];
#pragma unroll
    for (int cc = 0; cc < 8; ++cc) { gsc[cc] = gamma[c0 + cc] * rs; gbi[cc] = beta[c0 + cc]; }

    for (int n = threadIdx.x; n < N_DIM; n += 256) {
        v8h vals;
#pragma unroll
        for (int cc = 0; cc < 8; ++cc)
            vals[cc] = (_Float16)((xp[cc * N_DIM + n] - mu) * gsc[cc] + gbi[cc]);
        *(v8h*)(xh + ((size_t)b * N_DIM + n) * C_DIM + c0) = vals;
    }
}

// ---------------- Kernel 2: fused Q/K/V projection (WMMA, 16x64 per wave) ----------------
// q,k : [b][h][n][32]  (A-ready rows; K-columns contiguous)
// v   : [b][h][32][n]  (n contiguous so P*V B-fragments vector-load)
__global__ void qkv_kernel(const _Float16* __restrict__ xh,
                           const _Float16* __restrict__ hwq, const float* __restrict__ bq,
                           const _Float16* __restrict__ hwk, const float* __restrict__ bk,
                           const _Float16* __restrict__ hwv, const float* __restrict__ bv,
                           _Float16* __restrict__ q, _Float16* __restrict__ k,
                           _Float16* __restrict__ v) {
    int wg  = blockIdx.x * 8 + (threadIdx.x >> 5);  // 0..6143
    int m   = wg >> 11;                             // 0=q 1=k 2=v (2048 waves each)
    int rem = wg & 2047;
    int b   = rem >> 10;
    int t   = rem & 1023;
    int o0  = (t >> 6) * 16;
    int n0  = (t & 63) * 64;

    const _Float16* w   = (m == 0) ? hwq : (m == 1) ? hwk : hwv;
    const float*   bias = (m == 0) ? bq  : (m == 1) ? bk  : bv;
    float sc = (m < 2) ? ATT_SCALE : 1.0f;

    v8f acc0 = {}, acc1 = {}, acc2 = {}, acc3 = {};
#pragma unroll
    for (int c0 = 0; c0 < C_DIM; c0 += 32) {
        v16h a = load_frag_f16(w + o0 * C_DIM + c0, C_DIM);
        const _Float16* bbase = xh + ((size_t)b * N_DIM + n0) * C_DIM + c0;
        v16h bb0 = load_frag_f16(bbase,               C_DIM);
        v16h bb1 = load_frag_f16(bbase + 16 * C_DIM,  C_DIM);
        v16h bb2 = load_frag_f16(bbase + 32 * C_DIM,  C_DIM);
        v16h bb3 = load_frag_f16(bbase + 48 * C_DIM,  C_DIM);
        acc0 = WMMA_F16(a, bb0, acc0);
        acc1 = WMMA_F16(a, bb1, acc1);
        acc2 = WMMA_F16(a, bb2, acc2);
        acc3 = WMMA_F16(a, bb3, acc3);
    }

    int lane = threadIdx.x & 31;
    int gHalf = lane >> 4, col = lane & 15;
#pragma unroll
    for (int r = 0; r < 8; ++r) {
        int o  = o0 + r + gHalf * 8;
        int h  = o >> 5, dd = o & 31;
        float bval = bias[o];
        float v0 = (acc0[r] + bval) * sc;
        float v1 = (acc1[r] + bval) * sc;
        float v2 = (acc2[r] + bval) * sc;
        float v3 = (acc3[r] + bval) * sc;
        int na = n0 + col;
        if (m == 2) {
            size_t base = (((size_t)b * NUM_HEADS + h) * CH + dd) * N_DIM + na;
            v[base]      = (_Float16)v0;
            v[base + 16] = (_Float16)v1;
            v[base + 32] = (_Float16)v2;
            v[base + 48] = (_Float16)v3;
        } else {
            _Float16* dst = (m == 0) ? q : k;
            size_t base = (((size_t)b * NUM_HEADS + h) * N_DIM + na) * CH + dd;
            dst[base]           = (_Float16)v0;
            dst[base + 16 * CH] = (_Float16)v1;
            dst[base + 32 * CH] = (_Float16)v2;
            dst[base + 48 * CH] = (_Float16)v3;
        }
    }
}

// ---------------- Kernel 3: flash attention (one wave per 16-row tile) ----------------
__global__ void attn_kernel(const _Float16* __restrict__ q, const _Float16* __restrict__ kmat,
                            const _Float16* __restrict__ v, _Float16* __restrict__ attn) {
    __shared__ __align__(16) _Float16 ptile[4][16 * 32];   // per-wave P staging tile
    int wid = threadIdx.x >> 5;
    int wg = blockIdx.x * 4 + wid;           // 0..4095
    int b  = wg >> 11;
    int h  = (wg >> 8) & 7;
    int i0 = (wg & 255) * 16;

    size_t headoff = ((size_t)b * NUM_HEADS + h) * N_DIM;
    const _Float16* qp = q + (headoff + i0) * CH;
    const _Float16* vhead = v + ((size_t)b * NUM_HEADS + h) * CH * N_DIM; // [32][4096]
    v16h qa = load_frag_f16(qp, CH);         // Q rows are A-layout ready (K = d = 32)

    int lane = threadIdx.x & 31;
    int gHalf = lane >> 4, col = lane & 15;

    float m_[8], l_[8];
    v8f o0 = {}, o1 = {};
#pragma unroll
    for (int r = 0; r < 8; ++r) { m_[r] = -3.0e38f; l_[r] = 0.f; }

    _Float16* pt = ptile[wid];

    for (int j0 = 0; j0 < N_DIM; j0 += 32) {
        const _Float16* kp = kmat + (headoff + j0) * CH;
        __builtin_prefetch(kp + 32 * CH, 0, 1);            // next K chunk
        __builtin_prefetch(vhead + j0 + 32, 0, 1);         // next V columns

        // scores: S = Q^T K ; K columns (j) are contiguous 32-half rows -> vector loads
        v16h kb0 = load_frag_f16(kp, CH);
        v16h kb1 = load_frag_f16(kp + 16 * CH, CH);
        v8f s0 = {}, s1 = {};
        s0 = WMMA_F16(qa, kb0, s0);
        s1 = WMMA_F16(qa, kb1, s1);

        // online softmax: rows live across 16-lane groups in C layout
#pragma unroll
        for (int r = 0; r < 8; ++r) {
            float mx = fmaxf(s0[r], s1[r]);
            mx = fmaxf(mx, __shfl_xor(mx, 1, 32));
            mx = fmaxf(mx, __shfl_xor(mx, 2, 32));
            mx = fmaxf(mx, __shfl_xor(mx, 4, 32));
            mx = fmaxf(mx, __shfl_xor(mx, 8, 32));
            float mn = fmaxf(m_[r], mx);
            float alpha = __expf(m_[r] - mn);
            m_[r] = mn;
            float p0 = __expf(s0[r] - mn);
            float p1 = __expf(s1[r] - mn);
            float rsum = p0 + p1;
            rsum += __shfl_xor(rsum, 1, 32);
            rsum += __shfl_xor(rsum, 2, 32);
            rsum += __shfl_xor(rsum, 4, 32);
            rsum += __shfl_xor(rsum, 8, 32);
            l_[r] = l_[r] * alpha + rsum;
            o0[r] *= alpha;
            o1[r] *= alpha;
            int row = r + gHalf * 8;
            pt[row * 32 + col]      = (_Float16)p0;
            pt[row * 32 + col + 16] = (_Float16)p1;
        }
        __syncthreads();   // C-layout -> A-layout re-fragmentation through LDS

        v16h pa  = load_frag_f16(pt, 32);                  // 2x ds_load_b128 / lane
        // B[k=j][col=dd] = vhead[dd][j0+k] : dd-column contiguous in n -> vector loads
        v16h vb0 = load_frag_f16(vhead + j0, N_DIM);
        v16h vb1 = load_frag_f16(vhead + (size_t)16 * N_DIM + j0, N_DIM);
        o0 = WMMA_F16(pa, vb0, o0);
        o1 = WMMA_F16(pa, vb1, o1);
        __syncthreads();
    }

    // finalize: O / l ; store attn[b][n=i][c=h*32+dd] (c contiguous for out-proj B)
#pragma unroll
    for (int r = 0; r < 8; ++r) {
        int i = i0 + r + gHalf * 8;
        float inv = 1.0f / l_[r];
        size_t base = ((size_t)b * N_DIM + i) * C_DIM + h * CH;
        attn[base + col]      = (_Float16)(o0[r] * inv);
        attn[base + col + 16] = (_Float16)(o1[r] * inv);
    }
}

// ---------------- Kernel 4: output projection + bias + residual (16x64 per wave) ----------------
__global__ void proj_kernel(const _Float16* __restrict__ attn, const _Float16* __restrict__ hwo,
                            const float* __restrict__ bout, const float* __restrict__ x,
                            float* __restrict__ out) {
    int wg = blockIdx.x * 8 + (threadIdx.x >> 5);   // 0..2047
    int b  = wg >> 10;
    int t  = wg & 1023;
    int o0 = (t >> 6) * 16;
    int n0 = (t & 63) * 64;

    v8f acc0 = {}, acc1 = {}, acc2 = {}, acc3 = {};
#pragma unroll
    for (int c0 = 0; c0 < C_DIM; c0 += 32) {
        v16h a = load_frag_f16(hwo + o0 * C_DIM + c0, C_DIM);
        const _Float16* bbase = attn + ((size_t)b * N_DIM + n0) * C_DIM + c0;
        v16h bb0 = load_frag_f16(bbase,              C_DIM);
        v16h bb1 = load_frag_f16(bbase + 16 * C_DIM, C_DIM);
        v16h bb2 = load_frag_f16(bbase + 32 * C_DIM, C_DIM);
        v16h bb3 = load_frag_f16(bbase + 48 * C_DIM, C_DIM);
        acc0 = WMMA_F16(a, bb0, acc0);
        acc1 = WMMA_F16(a, bb1, acc1);
        acc2 = WMMA_F16(a, bb2, acc2);
        acc3 = WMMA_F16(a, bb3, acc3);
    }

    int lane = threadIdx.x & 31;
    int gHalf = lane >> 4, col = lane & 15;
#pragma unroll
    for (int r = 0; r < 8; ++r) {
        int o = o0 + r + gHalf * 8;
        float bval = bout[o];
        size_t idx = ((size_t)b * C_DIM + o) * N_DIM + n0 + col;
        out[idx]      = acc0[r] + bval + x[idx];
        out[idx + 16] = acc1[r] + bval + x[idx + 16];
        out[idx + 32] = acc2[r] + bval + x[idx + 32];
        out[idx + 48] = acc3[r] + bval + x[idx + 48];
    }
}

// ---------------- launch ----------------
extern "C" void kernel_launch(void* const* d_in, const int* in_sizes, int n_in,
                              void* d_out, int out_size, void* d_ws, size_t ws_size,
                              hipStream_t stream) {
    (void)in_sizes; (void)n_in; (void)out_size; (void)ws_size;
    const float* x     = (const float*)d_in[0];
    const float* gamma = (const float*)d_in[1];
    const float* beta  = (const float*)d_in[2];
    const float* wq    = (const float*)d_in[3];
    const float* bq    = (const float*)d_in[4];
    const float* wk    = (const float*)d_in[5];
    const float* bk    = (const float*)d_in[6];
    const float* wv    = (const float*)d_in[7];
    const float* bv    = (const float*)d_in[8];
    const float* wout  = (const float*)d_in[9];
    const float* bout  = (const float*)d_in[10];
    float* out = (float*)d_out;

    // workspace: 5 f16 activation buffers (4 MB each) + 4 f16 weight buffers (128 KB each)
    const size_t BUF  = (size_t)B_DIM * C_DIM * N_DIM * sizeof(_Float16);
    const size_t WBUF = (size_t)C_DIM * C_DIM * sizeof(_Float16);
    char* ws = (char*)d_ws;
    _Float16* xh   = (_Float16*)(ws + 0 * BUF);   // [b][n][c]
    _Float16* qb   = (_Float16*)(ws + 1 * BUF);   // [b][h][n][32]
    _Float16* kb   = (_Float16*)(ws + 2 * BUF);   // [b][h][n][32]
    _Float16* vb   = (_Float16*)(ws + 3 * BUF);   // [b][h][32][n]
    _Float16* attn = (_Float16*)(ws + 4 * BUF);   // [b][n][c]
    _Float16* hwq  = (_Float16*)(ws + 5 * BUF);
    _Float16* hwk  = (_Float16*)(ws + 5 * BUF + 1 * WBUF);
    _Float16* hwv  = (_Float16*)(ws + 5 * BUF + 2 * WBUF);
    _Float16* hwo  = (_Float16*)(ws + 5 * BUF + 3 * WBUF);

    wcvt_kernel<<<64, 256, 0, stream>>>(wq, wk, wv, wout, hwq, hwk, hwv, hwo);
    gn_kernel<<<B_DIM * 32, 256, 0, stream>>>(x, gamma, beta, xh);
    // 3 mats * 2 batches * 16 o-tiles * 64 n-tiles = 6144 waves / 8 per block
    qkv_kernel<<<768, 256, 0, stream>>>(xh, hwq, bq, hwk, bk, hwv, bv, qb, kb, vb);
    // 2 * 8 heads * 256 i-tiles = 4096 waves / 4 per block
    attn_kernel<<<1024, 128, 0, stream>>>(qb, kb, vb, attn);
    // 2 * 16 o-tiles * 64 n-tiles = 2048 waves / 8 per block
    proj_kernel<<<256, 256, 0, stream>>>(attn, hwo, bout, x, out);
}
